// Net_12799002542679
// MI455X (gfx1250) — matrix operand order repaired and checked by reading, not compiled, not run on hardware
//
#include <hip/hip_runtime.h>

#define DIM     64
#define NVAR    60000
#define NCON    40000
#define NNODE   100000
#define NEDGE   500000
#define NLAYER  4

typedef __attribute__((ext_vector_type(16))) __bf16 v16bf;
typedef __attribute__((ext_vector_type(8)))  float  v8f;

union F4B8 { float4 f; __bf16 h[8]; };

__device__ __forceinline__ v8f wmma_bf16(v16bf a, v16bf b, v8f c) {
    // D = A(16x32 bf16) * B(32x16 bf16) + C(16x16 f32)
    return __builtin_amdgcn_wmma_f32_16x16x32_bf16(
        /*neg_a=*/false, a, /*neg_b=*/false, b,
        /*c_mod=*/(short)0, c, /*reuse_a=*/false, /*reuse_b=*/false);
}

// Build bf16 A-fragment for one 32-wide K chunk from an f32 row (64 floats).
// Contiguous-K A layout (B loads match): lane holds row M=lane&15; elements 0..15 = K run
// b0..b0+15, b0 = kc*32 + (lane>=16)*16.
__device__ __forceinline__ v16bf a_from_f32row(const float* __restrict__ rowp, int b0) {
    const float4 f0 = *(const float4*)(rowp + b0);
    const float4 f1 = *(const float4*)(rowp + b0 + 4);
    const float4 f2 = *(const float4*)(rowp + b0 + 8);
    const float4 f3 = *(const float4*)(rowp + b0 + 12);
    v16bf a;
    a[0]=(__bf16)f0.x;  a[1]=(__bf16)f0.y;  a[2]=(__bf16)f0.z;  a[3]=(__bf16)f0.w;
    a[4]=(__bf16)f1.x;  a[5]=(__bf16)f1.y;  a[6]=(__bf16)f1.z;  a[7]=(__bf16)f1.w;
    a[8]=(__bf16)f2.x;  a[9]=(__bf16)f2.y;  a[10]=(__bf16)f2.z; a[11]=(__bf16)f2.w;
    a[12]=(__bf16)f3.x; a[13]=(__bf16)f3.y; a[14]=(__bf16)f3.z; a[15]=(__bf16)f3.w;
    return a;
}

// Same pattern but from a bf16 row (LDS activation tile in the head).
__device__ __forceinline__ v16bf a_from_bf16row(const __bf16* __restrict__ rowp, int b0) {
    F4B8 u0, u1;
    u0.f = *(const float4*)(rowp + b0);        // 8 bf16 = 16B
    u1.f = *(const float4*)(rowp + b0 + 8);
    v16bf a;
#pragma unroll
    for (int j = 0; j < 8; ++j) { a[j] = u0.h[j]; a[8 + j] = u1.h[j]; }
    return a;
}

// ---------------------------------------------------------------------------
// Zero scratch (vectorized; n must be a multiple of 4)
__global__ void __launch_bounds__(256) zero_kernel(float4* __restrict__ p, long n4) {
    long t = (long)blockIdx.x * blockDim.x + threadIdx.x;
    if (t < n4) p[t] = make_float4(0.f, 0.f, 0.f, 0.f);
}

// ---------------------------------------------------------------------------
// Pre-transpose weights to bf16 (contiguous K per output column)
__global__ void __launch_bounds__(256) prep_weights(
    const float* __restrict__ htv_w1, const float* __restrict__ w_cons,
    const float* __restrict__ w_vars, const float* __restrict__ fc1w,
    const float* __restrict__ fc2w,  const float* __restrict__ fc3w,
    __bf16* __restrict__ htv_w1T, __bf16* __restrict__ wcombT,
    __bf16* __restrict__ fc1T, __bf16* __restrict__ fc2T, __bf16* __restrict__ fc3T)
{
    int t = blockIdx.x * blockDim.x + threadIdx.x;
    if (t < 16384) {                       // htv_w1T[l][n][k] = htv_w1[l][k][n]
        int l = t >> 12, rem = t & 4095, n = rem >> 6, k = rem & 63;
        htv_w1T[t] = (__bf16)htv_w1[l * 4096 + k * 64 + n];
        return;
    }
    t -= 16384;
    if (t < 32768) {                       // W_comb[l][n][k], K=128: top=w_cons(pad), bottom=w_vars
        int l = t >> 13, rem = t & 8191, n = rem >> 7, k = rem & 127;
        float v;
        if (k < 64) {
            if (k < 62 && n < 62)        v = w_cons[l * 3844 + k * 62 + n];
            else if (k == 62 && n == 62) v = 1.0f;   // col-62 passthrough of s0[62]
            else                         v = 0.0f;
        } else {
            v = w_vars[l * 4096 + (k - 64) * 64 + n];
        }
        wcombT[l * 8192 + n * 128 + k] = (__bf16)v;
        return;
    }
    t -= 32768;
    if (t < 20480) { int n = t / 320, k = t % 320; fc1T[t] = (__bf16)fc1w[k * 64 + n]; return; }
    t -= 20480;
    if (t < 4096) { int n = t >> 6, k = t & 63; fc2T[t] = (__bf16)fc2w[k * 64 + n]; return; }
    t -= 4096;
    if (t < 4096) { int n = t >> 6, k = t & 63; fc3T[t] = (__bf16)fc3w[k * 64 + n]; }
}

// ---------------------------------------------------------------------------
// Initial node features via tiny MLPs. One thread per (node, out-col).
__global__ void __launch_bounds__(256) init_nodes(
    const float* __restrict__ feat, int count, const int* __restrict__ assoc,
    const float* __restrict__ w1, const float* __restrict__ b1,
    const float* __restrict__ w2, const float* __restrict__ b2,
    int W, float* __restrict__ x0)
{
    long t = (long)blockIdx.x * blockDim.x + threadIdx.x;
    if (t >= (long)count * 64) return;
    int i = (int)(t >> 6);
    int k = (int)(t & 63);
    float v = feat[i];
    float out;
    if (k < W) {
        float s = b2[k];
        for (int j = 0; j < W; ++j) {
            float h = fmaxf(v * w1[j] + b1[j], 0.0f);
            s += h * w2[j * W + k];
        }
        out = s;
    } else if (k == W) out = v;      // raw feature
    else               out = 1.0f;   // ones column (var nodes, W==62)
    x0[(long)assoc[i] * DIM + k] = out;
}

// ---------------------------------------------------------------------------
// Edge pass: scatter x[src] into s0/s1 (by edge type), WMMA sigmoid path -> va.
// Persistent waves: B-fragments + per-lane b1/w2 hoisted, grid-stride tiles.
__global__ void __launch_bounds__(256) edge_kernel(
    const float* __restrict__ x, const int* __restrict__ ei, const int* __restrict__ et,
    const float* __restrict__ ef, const __bf16* __restrict__ w1T,
    const float* __restrict__ b1, const float* __restrict__ w2,
    const float* __restrict__ b2p,
    float* __restrict__ s0, float* __restrict__ s1, float* __restrict__ va)
{
    const int lane = threadIdx.x & 31;
    const int wv   = threadIdx.x >> 5;
    const int row  = lane & 15;
    const int half = lane >> 4;
    const int wid    = blockIdx.x * 8 + wv;
    const int nwaves = gridDim.x * 8;
    const int nTiles = NEDGE / 16;

    // Hoisted weights (L2-hot, loaded once per wave)
    v16bf Bf[4][2];
    float b1v[4], w2v[4];
#pragma unroll
    for (int nc = 0; nc < 4; ++nc) {
        const int col = nc * 16 + row;
#pragma unroll
        for (int kc = 0; kc < 2; ++kc)
            Bf[nc][kc] = *(const v16bf*)(w1T + col * 64 + kc * 32 + half * 16);
        b1v[nc] = b1[col];
        w2v[nc] = w2[col];
    }
    const float b2v = b2p[0];

    for (int tile = wid; tile < nTiles; tile += nwaves) {
        const int e    = tile * 16 + row;
        const int srcv = ei[e];
        const int dstv = ei[NEDGE + e];
        const int etv  = et[e];
        const float* xrow = x + (long)srcv * DIM;
        float* sdst = (etv == 0 ? s0 : s1) + (long)dstv * DIM;

        // Gather row, scatter-add into s0/s1 (L2-resident atomics), A frags.
        v16bf a[2];
#pragma unroll
        for (int kc = 0; kc < 2; ++kc) {
            const int b0 = kc * 32 + half * 16;
            __align__(16) float tmp[16];
            *(float4*)&tmp[0]  = *(const float4*)(xrow + b0);
            *(float4*)&tmp[4]  = *(const float4*)(xrow + b0 + 4);
            *(float4*)&tmp[8]  = *(const float4*)(xrow + b0 + 8);
            *(float4*)&tmp[12] = *(const float4*)(xrow + b0 + 12);
#pragma unroll
            for (int j = 0; j < 16; ++j) {
                const int col = b0 + j;
                atomicAdd(&sdst[col], tmp[j]);
                a[kc][j] = (__bf16)tmp[j];
            }
        }

        // h = sigmoid(xj @ W1 + b1); per-row dot with w2 accumulated per lane.
        float part[8];
#pragma unroll
        for (int r = 0; r < 8; ++r) part[r] = 0.0f;

#pragma unroll
        for (int nc = 0; nc < 4; ++nc) {
            v8f acc = {0.f,0.f,0.f,0.f,0.f,0.f,0.f,0.f};
#pragma unroll
            for (int kc = 0; kc < 2; ++kc)
                acc = wmma_bf16(a[kc], Bf[nc][kc], acc);
#pragma unroll
            for (int r = 0; r < 8; ++r) {
                const float h = acc[r] + b1v[nc];
                part[r] += w2v[nc] * (1.0f / (1.0f + __expf(-h)));
            }
        }

        // Reduce across the 16-lane half (rows r+half*8 end in lanes 0 / 16).
#pragma unroll
        for (int m = 1; m < 16; m <<= 1) {
#pragma unroll
            for (int r = 0; r < 8; ++r) part[r] += __shfl_xor(part[r], m, 32);
        }

        if (row == 0) {
#pragma unroll
            for (int r = 0; r < 8; ++r) {
                const int e2 = tile * 16 + half * 8 + r;
                if (et[e2] == 0) {
                    const int s2 = ei[e2];
                    const int d2 = ei[NEDGE + e2];
                    atomicAdd(&va[d2], (part[r] + b2v) * ef[s2]);
                }
            }
        }
    }
}

// ---------------------------------------------------------------------------
// Node pass: x_out = relu([s0|s1] @ W_comb + e63*va + bias).
// Persistent waves with all 16 B-fragments hoisted.
__global__ void __launch_bounds__(256) node_kernel(
    const float* __restrict__ s0, const float* __restrict__ s1,
    const float* __restrict__ va, const __bf16* __restrict__ wT,
    const float* __restrict__ bias, float* __restrict__ xout)
{
    const int lane = threadIdx.x & 31;
    const int wv   = threadIdx.x >> 5;
    const int row  = lane & 15;
    const int half = lane >> 4;
    const int wid    = blockIdx.x * 8 + wv;
    const int nwaves = gridDim.x * 8;
    const int nTiles = NNODE / 16;

    v16bf Bf[4][4];
    float bv[4];
#pragma unroll
    for (int nc = 0; nc < 4; ++nc) {
        const int col = nc * 16 + row;
#pragma unroll
        for (int kc = 0; kc < 4; ++kc)
            Bf[nc][kc] = *(const v16bf*)(wT + col * 128 + kc * 32 + half * 16);
        bv[nc] = bias[col];
    }

    for (int tile = wid; tile < nTiles; tile += nwaves) {
        const long nb = (long)tile * 16;
        v16bf a[4];
#pragma unroll
        for (int kc = 0; kc < 4; ++kc) {
            const float* rp = (kc < 2 ? s0 : s1) + (nb + row) * DIM;
            a[kc] = a_from_f32row(rp, (kc & 1) * 32 + half * 16);
        }

#pragma unroll
        for (int nc = 0; nc < 4; ++nc) {
            const int col = nc * 16 + row;
            v8f acc = {0.f,0.f,0.f,0.f,0.f,0.f,0.f,0.f};
#pragma unroll
            for (int kc = 0; kc < 4; ++kc)
                acc = wmma_bf16(a[kc], Bf[nc][kc], acc);
#pragma unroll
            for (int r = 0; r < 8; ++r) {
                const long nm = nb + half * 8 + r;
                float v = acc[r] + bv[nc];
                if (col == 63) v += va[nm];
                xout[nm * DIM + col] = fmaxf(v, 0.0f);
            }
        }
    }
}

// ---------------------------------------------------------------------------
// Head: fc1(320->64) -> fc2 -> fc3 -> fc4(64->1). One wave per 16 vars,
// per-wave LDS bf16 activation tiles between stages.
__global__ void __launch_bounds__(256) head_kernel(
    const float* __restrict__ x, const int* __restrict__ assocv,
    const __bf16* __restrict__ fc1T, const float* __restrict__ fc1b,
    const __bf16* __restrict__ fc2T, const float* __restrict__ fc2b,
    const __bf16* __restrict__ fc3T, const float* __restrict__ fc3b,
    const float* __restrict__ fc4w,  const float* __restrict__ fc4b,
    float* __restrict__ out)
{
    __shared__ __align__(16) __bf16 acts[8][2][16 * DIM];
    const int lane = threadIdx.x & 31;
    const int wv   = threadIdx.x >> 5;
    const int tile = blockIdx.x * 8 + wv;
    if (tile >= NVAR / 16) return;
    const int row  = lane & 15;
    const int half = lane >> 4;
    const int v    = tile * 16 + row;
    const long node = assocv[v];

    __bf16* A0 = &acts[wv][0][0];
    __bf16* A1 = &acts[wv][1][0];

    // ---- fc1: K = 320 = 10 chunks of 32 spanning the 5 x-buffers
    v8f acc[4];
#pragma unroll
    for (int nc = 0; nc < 4; ++nc) acc[nc] = (v8f){0.f,0.f,0.f,0.f,0.f,0.f,0.f,0.f};
#pragma unroll
    for (int kc = 0; kc < 10; ++kc) {
        const int Kg = kc * 32;
        const float* xr = x + (size_t)(Kg / 64) * NNODE * DIM + node * DIM;
        const v16bf a = a_from_f32row(xr, (Kg & 63) + half * 16);
#pragma unroll
        for (int nc = 0; nc < 4; ++nc) {
            const int col = nc * 16 + row;
            const v16bf b = *(const v16bf*)(fc1T + col * 320 + kc * 32 + half * 16);
            acc[nc] = wmma_bf16(a, b, acc[nc]);
        }
    }
#pragma unroll
    for (int nc = 0; nc < 4; ++nc) {
        const int col = nc * 16 + row;
        const float bv = fc1b[col];
#pragma unroll
        for (int r = 0; r < 8; ++r)
            A0[(half * 8 + r) * DIM + col] = (__bf16)fmaxf(acc[nc][r] + bv, 0.0f);
    }

    // ---- fc2 (A from LDS buf0 -> buf1)
#pragma unroll
    for (int nc = 0; nc < 4; ++nc) acc[nc] = (v8f){0.f,0.f,0.f,0.f,0.f,0.f,0.f,0.f};
#pragma unroll
    for (int kc = 0; kc < 2; ++kc) {
        const v16bf a = a_from_bf16row(A0 + row * DIM, kc * 32 + half * 16);
#pragma unroll
        for (int nc = 0; nc < 4; ++nc) {
            const int col = nc * 16 + row;
            const v16bf b = *(const v16bf*)(fc2T + col * 64 + kc * 32 + half * 16);
            acc[nc] = wmma_bf16(a, b, acc[nc]);
        }
    }
#pragma unroll
    for (int nc = 0; nc < 4; ++nc) {
        const int col = nc * 16 + row;
        const float bv = fc2b[col];
#pragma unroll
        for (int r = 0; r < 8; ++r)
            A1[(half * 8 + r) * DIM + col] = (__bf16)fmaxf(acc[nc][r] + bv, 0.0f);
    }

    // ---- fc3 (buf1 -> buf0)
#pragma unroll
    for (int nc = 0; nc < 4; ++nc) acc[nc] = (v8f){0.f,0.f,0.f,0.f,0.f,0.f,0.f,0.f};
#pragma unroll
    for (int kc = 0; kc < 2; ++kc) {
        const v16bf a = a_from_bf16row(A1 + row * DIM, kc * 32 + half * 16);
#pragma unroll
        for (int nc = 0; nc < 4; ++nc) {
            const int col = nc * 16 + row;
            const v16bf b = *(const v16bf*)(fc3T + col * 64 + kc * 32 + half * 16);
            acc[nc] = wmma_bf16(a, b, acc[nc]);
        }
    }
#pragma unroll
    for (int nc = 0; nc < 4; ++nc) {
        const int col = nc * 16 + row;
        const float bv = fc3b[col];
#pragma unroll
        for (int r = 0; r < 8; ++r)
            A0[(half * 8 + r) * DIM + col] = (__bf16)fmaxf(acc[nc][r] + bv, 0.0f);
    }

    // ---- fc4: per-row dot, 2 lanes per row (halves of the 64 cols)
    float sum = 0.0f;
    const int cs = half * 32;
#pragma unroll
    for (int c = 0; c < 32; ++c)
        sum += (float)A0[row * DIM + cs + c] * fc4w[cs + c];
    sum += __shfl_xor(sum, 16, 32);
    if (half == 0) out[v] = sum + fc4b[0];
}

// ---------------------------------------------------------------------------
extern "C" void kernel_launch(void* const* d_in, const int* in_sizes, int n_in,
                              void* d_out, int out_size, void* d_ws, size_t ws_size,
                              hipStream_t stream)
{
    const float* varf   = (const float*)d_in[0];
    const float* conf   = (const float*)d_in[1];
    const float* ef     = (const float*)d_in[2];
    const int*   ei     = (const int*)d_in[3];
    const int*   et     = (const int*)d_in[4];
    const int*   assocv = (const int*)d_in[5];
    const int*   assocc = (const int*)d_in[6];
    const float* vmw1 = (const float*)d_in[7];
    const float* vmb1 = (const float*)d_in[8];
    const float* vmw2 = (const float*)d_in[9];
    const float* vmb2 = (const float*)d_in[10];
    const float* cmw1 = (const float*)d_in[11];
    const float* cmb1 = (const float*)d_in[12];
    const float* cmw2 = (const float*)d_in[13];
    const float* cmb2 = (const float*)d_in[14];
    const float* htv_w1 = (const float*)d_in[15];
    const float* htv_b1 = (const float*)d_in[16];
    const float* htv_w2 = (const float*)d_in[17];
    const float* htv_b2 = (const float*)d_in[18];
    const float* w_cons = (const float*)d_in[19];
    const float* w_vars = (const float*)d_in[20];
    const float* bias   = (const float*)d_in[21];
    const float* fc1w = (const float*)d_in[22];
    const float* fc1b = (const float*)d_in[23];
    const float* fc2w = (const float*)d_in[24];
    const float* fc2b = (const float*)d_in[25];
    const float* fc3w = (const float*)d_in[26];
    const float* fc3b = (const float*)d_in[27];
    const float* fc4w = (const float*)d_in[28];
    const float* fc4b = (const float*)d_in[29];

    // Workspace layout (f32 then bf16; base assumed >=256B aligned)
    float* x  = (float*)d_ws;                                // 5 * N * 64
    float* s0 = x  + (size_t)5 * NNODE * DIM;                // N * 64
    float* s1 = s0 + (size_t)NNODE * DIM;                    // N * 64
    float* va = s1 + (size_t)NNODE * DIM;                    // N
    __bf16* bw       = (__bf16*)(va + NNODE);                // 32B-aligned by construction
    __bf16* htv_w1T  = bw;                                   // 4*64*64
    __bf16* wcombT   = htv_w1T + 4 * 64 * 64;                // 4*64*128
    __bf16* fc1T     = wcombT  + 4 * 64 * 128;               // 64*320
    __bf16* fc2T     = fc1T + 64 * 320;                      // 64*64
    __bf16* fc3T     = fc2T + 64 * 64;                       // 64*64

    // 1) bf16 transposed weights
    prep_weights<<<(77824 + 255) / 256, 256, 0, stream>>>(
        htv_w1, w_cons, w_vars, fc1w, fc2w, fc3w,
        htv_w1T, wcombT, fc1T, fc2T, fc3T);

    // 2) x0 via input MLPs
    init_nodes<<<((long)NVAR * 64 + 255) / 256, 256, 0, stream>>>(
        varf, NVAR, assocv, vmw1, vmb1, vmw2, vmb2, 62, x);
    init_nodes<<<((long)NCON * 64 + 255) / 256, 256, 0, stream>>>(
        conf, NCON, assocc, cmw1, cmb1, cmw2, cmb2, 63, x);

    // 3) message-passing layers
    const long zn4 = ((long)2 * NNODE * DIM + NNODE) / 4;   // s0|s1|va contiguous
    for (int l = 0; l < NLAYER; ++l) {
        zero_kernel<<<(zn4 + 255) / 256, 256, 0, stream>>>((float4*)s0, zn4);
        edge_kernel<<<1024, 256, 0, stream>>>(
            x + (size_t)l * NNODE * DIM, ei, et, ef,
            htv_w1T + l * 4096, htv_b1 + l * 64, htv_w2 + l * 64, htv_b2 + l,
            s0, s1, va);
        node_kernel<<<512, 256, 0, stream>>>(
            s0, s1, va, wcombT + l * 8192, bias + l * 64,
            x + (size_t)(l + 1) * NNODE * DIM);
    }

    // 4) dense head over variable nodes
    head_kernel<<<(NVAR / 16 + 7) / 8, 256, 0, stream>>>(
        x, assocv, fc1T, fc1b, fc2T, fc2b, fc3T, fc3b, fc4w, fc4b,
        (float*)d_out);
}
